// Decoder_34376918237736
// MI455X (gfx1250) — matrix operand order repaired and checked by reading, not compiled, not run on hardware
//
#include <hip/hip_runtime.h>
#include <hip/hip_bf16.h>

typedef __attribute__((ext_vector_type(16))) __bf16 v16bf;
typedef __attribute__((ext_vector_type(8)))  float  v8f;

// ---------------- sizes ----------------
#define Bsz 512
#define Lsz 256
#define Esz 256
#define Hsz 256

// workspace layout (in __bf16 elements)
#define WG_OFF  ((size_t)0)                    // wGates [1024][512]  (Wi|Wh per row)
#define WQ_OFF  ((size_t)524288)               // wWq    [256][256]
#define WO_OFF  ((size_t)(WQ_OFF + 65536))     // wWo    [256][512]
#define CTX_OFF ((size_t)(WO_OFF + 131072))    // ctx    [512][256][256]

// output layout (float elements)
#define OUT_PTR 512
#define OUT_HF  (512 + 512*256)
#define OUT_CF  (512 + 2*512*256)

// ---------------- helpers ----------------
__device__ inline v8f vzero() {
  v8f z = {0.f,0.f,0.f,0.f,0.f,0.f,0.f,0.f};
  return z;
}

struct BF16Frag { uint4 lo, hi; };

__device__ inline v16bf frag_cast(uint4 lo, uint4 hi) {
  BF16Frag f; f.lo = lo; f.hi = hi;
  return __builtin_bit_cast(v16bf, f);
}

// A-fragment (16x32 bf16, MxK) from a row-major bf16 matrix (LDS or global).
// Lane l (<16): M=l, K = k0 + {0..15}; lane l (>=16): K = k0 + {16..31}.
__device__ inline v16bf loadA_bf16(const __bf16* base, int ld, int k0, int lane) {
  int row  = lane & 15;
  int half = lane >> 4;
  const __bf16* p = base + (size_t)row * ld + k0 + half * 16;
  uint4 lo = *(const uint4*)p;          // K = k0+half*16 .. +7
  uint4 hi = *(const uint4*)(p + 8);    // K = k0+half*16+8 .. +15
  return frag_cast(lo, hi);
}

// B-fragment (32x16 bf16, KxN) where B[k][n] = W[n][k], W row-major bf16 [N][ld].
// Lane: n = n0 + lane%16 ; K = k0 + (lane/16)*16 + e  (16 contiguous elements).
__device__ inline v16bf loadB_bf16(const __bf16* W, int ld, int n0, int k0, int lane) {
  int n  = n0 + (lane & 15);
  int kb = k0 + (lane >> 4) * 16;
  const __bf16* p = W + (size_t)n * ld + kb;
  uint4 lo = *(const uint4*)p;
  uint4 hi = *(const uint4*)(p + 8);
  return frag_cast(lo, hi);
}

// A-fragment built from fp32 row-major matrix (converts to bf16 in regs)
__device__ inline v16bf loadA_f32(const float* M, int ld, int m0, int k0, int lane) {
  int row  = m0 + (lane & 15);
  int half = lane >> 4;
  const float* p1 = M + (size_t)row * ld + k0 + half * 16;
  const float* p2 = p1 + 8;
  v16bf a;
#pragma unroll
  for (int i = 0; i < 8; ++i) { a[i] = (__bf16)p1[i]; a[8 + i] = (__bf16)p2[i]; }
  return a;
}

__device__ inline v8f wmma_bf16(v16bf a, v16bf b, v8f c) {
  return __builtin_amdgcn_wmma_f32_16x16x32_bf16(
      /*neg_a=*/false, a, /*neg_b=*/false, b,
      /*c_mod=*/(short)0, c, /*reuse_a=*/false, /*reuse_b=*/false);
}

__device__ inline float sigm_f(float x) {
  return 1.0f / (1.0f + __expf(-x));
}
__device__ inline float tanh_f(float x) {
  float e = __expf(-2.0f * fabsf(x));
  float t = (1.0f - e) / (1.0f + e);
  return copysignf(t, x);
}

// ---------------- prep: fp32 weights -> bf16 workspace ----------------
__global__ void k_convert(const float* Wi, const float* Wh, const float* Wq,
                          const float* Wo, __bf16* ws) {
  int i = blockIdx.x * blockDim.x + threadIdx.x;
  if (i < 524288) {                       // wGates[n][k], k<256 -> Wi, else Wh
    int n = i >> 9, k = i & 511;
    float v = (k < 256) ? Wi[(size_t)n * 256 + k] : Wh[(size_t)n * 256 + (k - 256)];
    ws[WG_OFF + i] = (__bf16)v;
  } else if (i < 524288 + 65536) {
    int j = i - 524288;
    ws[WQ_OFF + j] = (__bf16)Wq[j];
  } else if (i < 524288 + 65536 + 131072) {
    int j = i - 524288 - 65536;
    ws[WO_OFF + j] = (__bf16)Wo[j];
  }
}

// ---------------- prep: ctx_proj[b][o][l] = Wc @ context[b]^T + bc ----------
__global__ void __launch_bounds__(256) k_ctx(const float* context, const float* Wc,
                                             const float* bc, __bf16* ctx_out) {
  int b    = blockIdx.x;
  int wave = threadIdx.x >> 5;
  int lane = threadIdx.x & 31;
  int tileId = blockIdx.y * 8 + wave;     // 0..255
  int ot = tileId >> 4, lt = tileId & 15;

  v8f acc = vzero();
  for (int kk = 0; kk < 8; ++kk) {
    int k0 = kk * 32;
    v16bf a = loadA_f32(Wc, Hsz, ot * 16, k0, lane);   // A[o][h]
    // B[k=h][n=l] = context[b][l][h] : 16 contiguous h per lane
    int l  = lt * 16 + (lane & 15);
    int kb = k0 + (lane >> 4) * 16;
    const float* p = context + ((size_t)b * Lsz + l) * Hsz + kb;
    v16bf bf;
#pragma unroll
    for (int i = 0; i < 16; ++i) bf[i] = (__bf16)p[i];
    acc = wmma_bf16(a, bf, acc);
  }
  int n = lt * 16 + (lane & 15);
#pragma unroll
  for (int r = 0; r < 8; ++r) {
    int m = ot * 16 + ((lane >> 4) ? 8 : 0) + r;
    float v = acc[r] + bc[m];
    ctx_out[((size_t)b * Hsz + m) * Lsz + n] = (__bf16)v;
  }
}

// ---------------- main persistent decoder ----------------
__global__ void __launch_bounds__(256, 1)
k_decode(const float* emb, const float* dec_in, const float* h0, const float* c0,
         const float* bi, const float* bh, const float* bo, const float* bq,
         const float* V, const __bf16* wGates, const __bf16* wWq,
         const __bf16* wWo, const __bf16* ctx, float* out) {
  __shared__ __bf16 A_xh[16][512];   // [x_t | h_{t-1}]  (gates GEMM A)
  __shared__ __bf16 A_o [16][512];   // [hid_t | hs]     (out GEMM A; hs half also q GEMM A)
  __shared__ float  q_s  [16][256];
  __shared__ float  att_s[16][256];
  __shared__ float  mask_s[16][256];
  __shared__ float  bias_ih[1024];
  __shared__ float  bq_s[256], bo_s[256], V_s[256];
  __shared__ float  outp_s[16];
  __shared__ int    idx_s[16];

  const int tid  = threadIdx.x;
  const int lane = tid & 31;
  const int w    = tid >> 5;          // wave id 0..7
  const int bg0  = blockIdx.x * 16;   // first batch row of tile
  const int nl   = lane & 15;
  const int mh   = (lane >> 4) ? 8 : 0;

  // ---- init ----
  for (int i = tid; i < 1024; i += 256) bias_ih[i] = bi[i] + bh[i];
  bq_s[tid] = bq[tid]; bo_s[tid] = bo[tid]; V_s[tid] = V[tid];
  if (tid < 16) outp_s[tid] = 1.0f;
#pragma unroll
  for (int b = 0; b < 16; ++b) {
    mask_s[b][tid] = 1.0f;
    A_xh[b][tid]       = (__bf16)dec_in[(size_t)(bg0 + b) * Esz + tid];
    A_xh[b][256 + tid] = (__bf16)h0[(size_t)(bg0 + b) * Hsz + tid];
  }
  float cfr[2][8];     // cell state fragments, resident in registers
#pragma unroll
  for (int p = 0; p < 2; ++p) {
    int j = (w + 8 * p) * 16 + nl;
#pragma unroll
    for (int r = 0; r < 8; ++r)
      cfr[p][r] = c0[(size_t)(bg0 + mh + r) * Hsz + j];
  }
  __syncthreads();

#pragma unroll 1
  for (int t = 0; t < 256; ++t) {
    // ---- Phase A: gates = [x|h] @ [Wi|Wh]^T ; wave w owns N-tiles {w+8s} ----
    v8f acc[8];
#pragma unroll
    for (int p = 0; p < 8; ++p) acc[p] = vzero();
    for (int kk = 0; kk < 16; ++kk) {
      int k0 = kk * 32;
      v16bf a = loadA_bf16(&A_xh[0][0], 512, k0, lane);
#pragma unroll
      for (int p = 0; p < 8; ++p) {
        v16bf bm = loadB_bf16(wGates, 512, (w + 8 * p) * 16, k0, lane);
        acc[p] = wmma_bf16(a, bm, acc[p]);
      }
    }
#pragma unroll
    for (int p = 0; p < 8; ++p) {
      float bv = bias_ih[(w + 8 * p) * 16 + nl];
#pragma unroll
      for (int r = 0; r < 8; ++r) acc[p][r] += bv;
    }
    // LSTM cell: acc[0..1]=i, [2..3]=f, [4..5]=g, [6..7]=o for j-tiles {w, w+8}
#pragma unroll
    for (int p2 = 0; p2 < 2; ++p2) {
      int j = (w + 8 * p2) * 16 + nl;
#pragma unroll
      for (int r = 0; r < 8; ++r) {
        int m = mh + r;
        float ig = sigm_f(acc[p2][r]);
        float fg = sigm_f(acc[2 + p2][r]);
        float gg = tanh_f(acc[4 + p2][r]);
        float og = sigm_f(acc[6 + p2][r]);
        float cn = fg * cfr[p2][r] + ig * gg;
        cfr[p2][r] = cn;
        float hs = og * tanh_f(cn);
        A_o[m][256 + j] = (__bf16)hs;
        if (t == 255) out[OUT_CF + (size_t)(bg0 + m) * Hsz + j] = cn;
      }
    }
    __syncthreads();

    // ---- Phase B: q = hs @ Wq^T + bq ----
    v8f qa[2]; qa[0] = vzero(); qa[1] = vzero();
    for (int kk = 0; kk < 8; ++kk) {
      int k0 = kk * 32;
      v16bf a = loadA_bf16(&A_o[0][0], 512, 256 + k0, lane);
#pragma unroll
      for (int p = 0; p < 2; ++p) {
        v16bf bm = loadB_bf16(wWq, 256, (w + 8 * p) * 16, k0, lane);
        qa[p] = wmma_bf16(a, bm, qa[p]);
      }
    }
#pragma unroll
    for (int p = 0; p < 2; ++p) {
      int n = (w + 8 * p) * 16 + nl;
      float bv = bq_s[n];
#pragma unroll
      for (int r = 0; r < 8; ++r) q_s[mh + r][n] = qa[p][r] + bv;
    }
    __syncthreads();

    // ---- Phase C: att[b][l] = sum_h V[h]*tanh(q[b][h] + ctx[b][h][l]) ----
    {
      int l = tid;
      for (int b = 0; b < 16; ++b) {
        const __bf16* cp = ctx + ((size_t)(bg0 + b) * Hsz) * Lsz + l;
        float s = 0.f;
#pragma unroll 4
        for (int h = 0; h < 256; ++h) {
          float cv = (float)cp[(size_t)h * Lsz];
          s += V_s[h] * tanh_f(q_s[b][h] + cv);
        }
        att_s[b][l] = (mask_s[b][l] == 0.0f) ? -INFINITY : s;
      }
    }
    __syncthreads();

    // ---- Phase D: softmax + top-1 (wave w handles rows w, w+8) ----
#pragma unroll
    for (int rr = 0; rr < 2; ++rr) {
      int b = w + 8 * rr;
      float mx = -INFINITY;
      for (int l = lane; l < 256; l += 32) mx = fmaxf(mx, att_s[b][l]);
#pragma unroll
      for (int off = 16; off > 0; off >>= 1) mx = fmaxf(mx, __shfl_xor(mx, off, 32));
      float sum = 0.f, best = -1.f; int bix = 1 << 30;
      for (int l = lane; l < 256; l += 32) {
        float e = __expf(att_s[b][l] - mx);
        att_s[b][l] = e;
        sum += e;
        if (e > best || (e == best && l < bix)) { best = e; bix = l; }
      }
#pragma unroll
      for (int off = 16; off > 0; off >>= 1) sum += __shfl_xor(sum, off, 32);
#pragma unroll
      for (int off = 16; off > 0; off >>= 1) {
        float ob = __shfl_xor(best, off, 32);
        int   oi = __shfl_xor(bix,  off, 32);
        if (ob > best || (ob == best && oi < bix)) { best = ob; bix = oi; }
      }
      float inv = 1.0f / sum;
      for (int l = lane; l < 256; l += 32) att_s[b][l] *= inv;   // alpha
      if (lane == 0) {
        outp_s[b] *= best * inv;
        mask_s[b][bix] = 0.0f;
        idx_s[b] = bix;
        out[OUT_PTR + (size_t)(bg0 + b) * Lsz + t] = (float)bix;
      }
    }
    __syncthreads();

    // ---- Phase E: hid_t[b][h] = sum_l ctx[b][h][l]*alpha[b][l] ; gather x ----
    {
      int h = tid;
      for (int b = 0; b < 16; ++b) {
        const __bf16* cp = ctx + ((size_t)(bg0 + b) * Hsz + h) * Lsz;
        float s = 0.f;
#pragma unroll 2
        for (int l0 = 0; l0 < 256; l0 += 8) {
          union { uint4 u; __bf16 e[8]; } cu;
          cu.u = *(const uint4*)(cp + l0);                 // 8 bf16, one b128 load
          float4 a0 = *(const float4*)&att_s[b][l0];
          float4 a1 = *(const float4*)&att_s[b][l0 + 4];
          s += (float)cu.e[0] * a0.x + (float)cu.e[1] * a0.y +
               (float)cu.e[2] * a0.z + (float)cu.e[3] * a0.w +
               (float)cu.e[4] * a1.x + (float)cu.e[5] * a1.y +
               (float)cu.e[6] * a1.z + (float)cu.e[7] * a1.w;
        }
        A_o[b][h] = (__bf16)s;
        A_xh[b][tid] =
            (__bf16)emb[((size_t)(bg0 + b) * Lsz + idx_s[b]) * Esz + tid];
      }
    }
    __syncthreads();

    // ---- Phase F: h_new = tanh([hid_t|hs] @ Wo^T + bo) ----
    v8f ha[2]; ha[0] = vzero(); ha[1] = vzero();
    for (int kk = 0; kk < 16; ++kk) {
      int k0 = kk * 32;
      v16bf a = loadA_bf16(&A_o[0][0], 512, k0, lane);
#pragma unroll
      for (int p = 0; p < 2; ++p) {
        v16bf bm = loadB_bf16(wWo, 512, (w + 8 * p) * 16, k0, lane);
        ha[p] = wmma_bf16(a, bm, ha[p]);
      }
    }
#pragma unroll
    for (int p = 0; p < 2; ++p) {
      int n = (w + 8 * p) * 16 + nl;
      float bv = bo_s[n];
#pragma unroll
      for (int r = 0; r < 8; ++r) {
        int m = mh + r;
        float hn = tanh_f(ha[p][r] + bv);
        A_xh[m][256 + n] = (__bf16)hn;
        if (t == 255) out[OUT_HF + (size_t)(bg0 + m) * Hsz + n] = hn;
      }
    }
    __syncthreads();
  }

  if (tid < 16) out[bg0 + tid] = outp_s[tid];
}

// ---------------- launch ----------------
extern "C" void kernel_launch(void* const* d_in, const int* in_sizes, int n_in,
                              void* d_out, int out_size, void* d_ws, size_t ws_size,
                              hipStream_t stream) {
  const float* emb    = (const float*)d_in[0];
  const float* dec_in = (const float*)d_in[1];
  const float* h0     = (const float*)d_in[2];
  const float* c0     = (const float*)d_in[3];
  const float* ctx_in = (const float*)d_in[4];
  const float* Wi     = (const float*)d_in[5];
  const float* bi     = (const float*)d_in[6];
  const float* Wh     = (const float*)d_in[7];
  const float* bh     = (const float*)d_in[8];
  const float* Wo     = (const float*)d_in[9];
  const float* bo     = (const float*)d_in[10];
  const float* Wq     = (const float*)d_in[11];
  const float* bq     = (const float*)d_in[12];
  const float* Wc     = (const float*)d_in[13];
  const float* bc     = (const float*)d_in[14];
  const float* V      = (const float*)d_in[15];
  float* out = (float*)d_out;

  __bf16* ws = (__bf16*)d_ws;
  __bf16* wGates = ws + WG_OFF;
  __bf16* wWq    = ws + WQ_OFF;
  __bf16* wWo    = ws + WO_OFF;
  __bf16* ctxp   = ws + CTX_OFF;

  // 1) weights -> bf16
  {
    int total = 524288 + 65536 + 131072;
    k_convert<<<(total + 255) / 256, 256, 0, stream>>>(Wi, Wh, Wq, Wo, ws);
  }
  // 2) ctx_proj (WMMA batched GEMM)
  {
    dim3 grid(Bsz, 32);
    k_ctx<<<grid, 256, 0, stream>>>(ctx_in, Wc, bc, ctxp);
  }
  // 3) persistent decoder: 32 workgroups x 256 threads, full 256-step scan
  k_decode<<<Bsz / 16, 256, 0, stream>>>(emb, dec_in, h0, c0, bi, bh, bo, bq, V,
                                         wGates, wWq, wWo, ctxp, out);
}
